// WubuBlock_5068061409874
// MI455X (gfx1250) — matrix operand order repaired and checked by reading, not compiled, not run on hardware
//
#include <hip/hip_runtime.h>
#include <hip/hip_bf16.h>

#define B_    2
#define N_    1024
#define DIM_  512
#define NH_   8
#define HD_   64
#define WIN_  64
#define FF_   2048
#define M_    (B_*N_)
#define EPS_  1e-7f

typedef __attribute__((ext_vector_type(16))) _Float16 v16h;
typedef __attribute__((ext_vector_type(8)))  _Float16 v8h;
typedef __attribute__((ext_vector_type(8)))  float    v8f;

__device__ __forceinline__ float gelu_f(float x) {
    const float k0 = 0.7978845608028654f; // sqrt(2/pi)
    float x3 = x * x * x;
    return 0.5f * x * (1.0f + tanhf(k0 * (x + 0.044715f * x3)));
}

// ---------------------------------------------------------------------------
// Transpose + f32->f16 weight conversion: Wt[nc][k] = (half)W[k][nc]
// ---------------------------------------------------------------------------
__global__ void convert_w_kernel(const float* __restrict__ W,
                                 _Float16* __restrict__ Wt, int K, int Nc) {
    int id = blockIdx.x * blockDim.x + threadIdx.x;
    if (id >= K * Nc) return;
    int nc = id / K;
    int k  = id % K;
    Wt[id] = (_Float16)W[(size_t)k * Nc + nc];
}

// ---------------------------------------------------------------------------
// LayerNorm over DIM_=512, one row per block (256 threads, 2 elems/thread),
// writes f16 for the following WMMA GEMM.
// ---------------------------------------------------------------------------
__global__ void layernorm_kernel(const float* __restrict__ x,
                                 const float* __restrict__ s,
                                 const float* __restrict__ bb,
                                 _Float16* __restrict__ out) {
    __shared__ float red[2][8];
    int row = blockIdx.x;
    const float* xr = x + (size_t)row * DIM_;
    float v0 = xr[threadIdx.x];
    float v1 = xr[threadIdx.x + 256];
    float sum = v0 + v1;
    float sq  = v0 * v0 + v1 * v1;
    for (int m = 16; m > 0; m >>= 1) {
        sum += __shfl_xor(sum, m, 32);
        sq  += __shfl_xor(sq,  m, 32);
    }
    int wave = threadIdx.x >> 5, lane = threadIdx.x & 31;
    if (lane == 0) { red[0][wave] = sum; red[1][wave] = sq; }
    __syncthreads();
    if (wave == 0) {
        float s0 = lane < 8 ? red[0][lane] : 0.f;
        float s1 = lane < 8 ? red[1][lane] : 0.f;
        for (int m = 4; m > 0; m >>= 1) {
            s0 += __shfl_xor(s0, m, 32);
            s1 += __shfl_xor(s1, m, 32);
        }
        if (lane == 0) { red[0][0] = s0; red[1][0] = s1; }
    }
    __syncthreads();
    float mean = red[0][0] * (1.0f / DIM_);
    float var  = red[1][0] * (1.0f / DIM_) - mean * mean;
    float inv  = rsqrtf(var + 1e-6f);
    size_t base = (size_t)row * DIM_;
    out[base + threadIdx.x]       = (_Float16)((v0 - mean) * inv * s[threadIdx.x]       + bb[threadIdx.x]);
    out[base + threadIdx.x + 256] = (_Float16)((v1 - mean) * inv * s[threadIdx.x + 256] + bb[threadIdx.x + 256]);
}

// ---------------------------------------------------------------------------
// WMMA GEMM, LDS-staged + async double buffering.
//   C[M][N] = A[M][K] (f16 row-major) @ Bt[N][K]^T (f16) + bias
// Block = 256 threads (8 waves) -> 128x64 output tile.
// Wave w: rows 16w..16w+15, 4 accumulators over the 64 block columns,
// reusing one A fragment per K-step (4 WMMAs back-to-back).
// B tile (64 cols x 32 K halves = 4KB) staged to LDS per K-step with
// global_load_async_to_lds_b128 (ASYNCcnt), double-buffered so the next
// tile streams in while WMMAs consume the current one.
// mode 0: Cf = acc+bias | mode 1: Ch = gelu(acc+bias) | mode 2: Cf = res+acc+bias
// ---------------------------------------------------------------------------
__global__ void wmma_gemm_kernel(const _Float16* __restrict__ A,
                                 const _Float16* __restrict__ Bt,
                                 const float* __restrict__ bias,
                                 float* __restrict__ Cf,
                                 _Float16* __restrict__ Ch,
                                 const float* __restrict__ res,
                                 int Ndim, int Kdim, int mode) {
    __shared__ __align__(32) _Float16 sb[2][64 * 32];   // [buf][col*32 + k]

    const int tid  = threadIdx.x;
    const int lane = tid & 31;
    const int wave = tid >> 5;
    const int nb   = Ndim >> 6;
    const int bm   = blockIdx.x / nb;
    const int bn   = blockIdx.x % nb;
    const int rowBase = bm << 7;
    const int colBase = bn << 6;

    const int l15 = lane & 15;
    const int hiA = (lane >> 4) << 3;    // A frag: added in Ap and per load -> contiguous K 0..15 / 16..31
    const int hiB = (lane >> 4) << 4;    // B frag: contiguous K 0..15 / 16..31

    // staging: each thread moves one 16B chunk (256 thr * 16B = 4KB tile)
    const int scol = tid >> 2;           // 0..63
    const int skc  = tid & 3;            // 0..3 (8-half chunks)
    const _Float16* gB = Bt + (size_t)(colBase + scol) * Kdim + skc * 8;
    const unsigned ldsOff   = (unsigned)(scol * 64 + skc * 16);     // bytes
    const unsigned ldsBase0 = (unsigned)(unsigned long long)(&sb[0][0]);
    const unsigned ldsBase1 = (unsigned)(unsigned long long)(&sb[1][0]);

    const _Float16* Ap = A + (size_t)(rowBase + (wave << 4) + l15) * Kdim + hiA;

    v8f acc0 = {}, acc1 = {}, acc2 = {}, acc3 = {};

    // prologue: stage K-step 0 into buffer 0
    asm volatile("global_load_async_to_lds_b128 %0, %1, off"
                 :: "v"(ldsBase0 + ldsOff), "v"(gB) : "memory");

    int cur = 0;
    for (int k0 = 0; k0 < Kdim; k0 += 32) {
        asm volatile("s_wait_asynccnt 0x0" ::: "memory");
        __syncthreads();                              // current buffer visible to all waves
        if (k0 + 32 < Kdim) {                         // prefetch next tile into other buffer
            unsigned dst = (cur ? ldsBase0 : ldsBase1) + ldsOff;
            asm volatile("global_load_async_to_lds_b128 %0, %1, off"
                         :: "v"(dst), "v"(gB + k0 + 32) : "memory");
        }
        // A fragment (per-wave rows, direct from global)
        v8h a0 = *(const v8h*)(Ap + k0 + hiA);
        v8h a1 = *(const v8h*)(Ap + k0 + hiA + 8);
        v16h a;
#pragma unroll
        for (int j = 0; j < 8; ++j) { a[j] = a0[j]; a[j + 8] = a1[j]; }
        // 4 B fragments from LDS
        const _Float16* sbc = &sb[cur][0];
        v16h b0 = *(const v16h*)(sbc + (0 * 16 + l15) * 32 + hiB);
        v16h b1 = *(const v16h*)(sbc + (1 * 16 + l15) * 32 + hiB);
        v16h b2 = *(const v16h*)(sbc + (2 * 16 + l15) * 32 + hiB);
        v16h b3 = *(const v16h*)(sbc + (3 * 16 + l15) * 32 + hiB);
        acc0 = __builtin_amdgcn_wmma_f32_16x16x32_f16(false, a, false, b0, (short)0, acc0, false, false);
        acc1 = __builtin_amdgcn_wmma_f32_16x16x32_f16(false, a, false, b1, (short)0, acc1, false, false);
        acc2 = __builtin_amdgcn_wmma_f32_16x16x32_f16(false, a, false, b2, (short)0, acc2, false, false);
        acc3 = __builtin_amdgcn_wmma_f32_16x16x32_f16(false, a, false, b3, (short)0, acc3, false, false);
        __syncthreads();                              // everyone done reading buf[cur]
        cur ^= 1;
    }

    // epilogue: D layout -> row m = rowBase + 16*wave + hiA + i, col = l15 + 16*t
    const int mbase = rowBase + (wave << 4) + hiA;
#pragma unroll
    for (int t = 0; t < 4; ++t) {
        v8f acc = (t == 0) ? acc0 : (t == 1) ? acc1 : (t == 2) ? acc2 : acc3;
        int col = colBase + t * 16 + l15;
        float bn_ = bias ? bias[col] : 0.0f;
#pragma unroll
        for (int i = 0; i < 8; ++i) {
            size_t off2 = (size_t)(mbase + i) * Ndim + col;
            float v = acc[i] + bn_;
            if (mode == 0)      Cf[off2] = v;
            else if (mode == 1) Ch[off2] = (_Float16)gelu_f(v);
            else                Cf[off2] = res[off2] + v;
        }
    }
}

// ---------------------------------------------------------------------------
// Post-projection: q += hash*log(9); RoPE(q,k); expmap0 into Poincare ball;
// emit ||q_h||^2 and ||k_h||^2. One wave per (b,n,h) 64-dim vector.
// ---------------------------------------------------------------------------
__global__ void qkv_post_kernel(float* __restrict__ q, float* __restrict__ k,
                                const float* __restrict__ cosb,
                                const float* __restrict__ sinb,
                                const float* __restrict__ cvec,
                                const float* __restrict__ hash,
                                float* __restrict__ qnorm,
                                float* __restrict__ knorm) {
    int wave = threadIdx.x >> 5;
    int lane = threadIdx.x & 31;
    int vec  = blockIdx.x * (blockDim.x >> 5) + wave;   // (b*N+n)*NH + h
    if (vec >= B_ * N_ * NH_) return;
    int h  = vec % NH_;
    int bn = vec / NH_;
    int n  = bn % N_;
    int b  = bn / N_;

    float c      = cvec[b];
    float sqrt_c = fmaxf(sqrtf(c), EPS_);
    float cs = cosb[n * (HD_ / 2) + lane];
    float sn = sinb[n * (HD_ / 2) + lane];
    size_t base = (size_t)vec * HD_;
    const float L9 = 2.1972245773362196f;   // log(1 + HASH_WIN)

    // ---- Q: hash offset + rope + expmap0 ----
    {
        float xr = q[base + lane]      + hash[h * HD_ + lane]      * L9;
        float xi = q[base + lane + 32] + hash[h * HD_ + lane + 32] * L9;
        float rr = xr * cs - xi * sn;
        float ri = xr * sn + xi * cs;
        float n2 = rr * rr + ri * ri;
        for (int m = 16; m > 0; m >>= 1) n2 += __shfl_xor(n2, m, 32);
        float nrm = sqrtf(n2);
        float f = 0.0f;
        if (nrm >= EPS_) {
            float safe = fmaxf(nrm, EPS_);
            f = tanhf(sqrt_c * safe) / (sqrt_c * safe);
        }
        q[base + lane]      = rr * f;
        q[base + lane + 32] = ri * f;
        if (lane == 0) qnorm[vec] = f * f * n2;
    }
    // ---- K: rope + expmap0 ----
    {
        float xr = k[base + lane];
        float xi = k[base + lane + 32];
        float rr = xr * cs - xi * sn;
        float ri = xr * sn + xi * cs;
        float n2 = rr * rr + ri * ri;
        for (int m = 16; m > 0; m >>= 1) n2 += __shfl_xor(n2, m, 32);
        float nrm = sqrtf(n2);
        float f = 0.0f;
        if (nrm >= EPS_) {
            float safe = fmaxf(nrm, EPS_);
            f = tanhf(sqrt_c * safe) / (sqrt_c * safe);
        }
        k[base + lane]      = rr * f;
        k[base + lane + 32] = ri * f;
        if (lane == 0) knorm[vec] = f * f * n2;
    }
}

// ---------------------------------------------------------------------------
// Sliding-window Poincare attention. 64 threads per query (4 queries/block).
// Scalar distance from (x2, y2, <x,y>); butterfly softmax in LDS; f16 output.
// ---------------------------------------------------------------------------
__global__ void attention_kernel(const float* __restrict__ qh,
                                 const float* __restrict__ kh,
                                 const float* __restrict__ v,
                                 const float* __restrict__ qn,
                                 const float* __restrict__ kn,
                                 const float* __restrict__ cvec,
                                 const float* __restrict__ geo,
                                 _Float16* __restrict__ attn) {
    __shared__ float sq[4][64];
    __shared__ float sw[4][64];
    int g = threadIdx.x >> 6;
    int w = threadIdx.x & 63;
    int qid = blockIdx.x * 4 + g;        // (b*NH + h)*N + n
    int n  = qid % N_;
    int bh = qid / N_;
    int h  = bh % NH_;
    int b  = bh / NH_;

    float c      = cvec[b];
    float sqrt_c = fmaxf(sqrtf(c), EPS_);
    size_t qbase = (((size_t)(b * N_ + n)) * NH_ + h) * HD_;
    sq[g][w] = qh[qbase + w];
    __syncthreads();

    float x2 = qn[(b * N_ + n) * NH_ + h];
    int idx = n + w - (WIN_ - 1);
    float xy = 0.0f, y2 = 0.0f;
    if (idx >= 0) {
        size_t kbase = (((size_t)(b * N_ + idx)) * NH_ + h) * HD_;
#pragma unroll 8
        for (int d = 0; d < HD_; ++d) xy += sq[g][d] * kh[kbase + d];
        y2 = kn[(b * N_ + idx) * NH_ + h];
    }

    // |alpha*x - beta*y|^2 expanded in scalars
    float alpha = 1.0f - 2.0f * c * xy + c * y2;
    float beta  = 1.0f - c * x2;
    float num2  = fmaxf(alpha * alpha * x2 - 2.0f * alpha * beta * xy + beta * beta * y2, 0.0f);
    float den   = fmaxf(1.0f - 2.0f * c * xy + c * c * x2 * y2, EPS_);
    float dn    = sqrtf(num2) / den;
    float arg   = fminf(sqrt_c * fmaxf(dn, EPS_), 1.0f - EPS_);
    float dist  = 2.0f * atanhf(arg) / sqrt_c;
    float score = -geo[h] * dist;

    // softmax over 64 window slots (butterfly, spans 2 waves -> LDS)
    sw[g][w] = score;
    __syncthreads();
    for (int m = 32; m > 0; m >>= 1) {
        float a_ = sw[g][w], b_ = sw[g][w ^ m];
        __syncthreads();
        sw[g][w] = fmaxf(a_, b_);
        __syncthreads();
    }
    float mx = sw[g][w];
    float e  = expf(score - mx);
    __syncthreads();
    sw[g][w] = e;
    __syncthreads();
    for (int m = 32; m > 0; m >>= 1) {
        float a_ = sw[g][w], b_ = sw[g][w ^ m];
        __syncthreads();
        sw[g][w] = a_ + b_;
        __syncthreads();
    }
    float wgt = e / sw[g][w];
    __syncthreads();
    sw[g][w] = wgt;
    __syncthreads();

    // thread w now plays dim d = w: out[d] = sum_w wgt[w] * v[key(w)][d]
    int d = w;
    float out = 0.0f;
    for (int ww = 0; ww < WIN_; ++ww) {
        int ki = n + ww - (WIN_ - 1);
        if (ki >= 0) {
            out += sw[g][ww] * v[(((size_t)(b * N_ + ki)) * NH_ + h) * HD_ + d];
        }
    }
    attn[((size_t)(b * N_ + n)) * DIM_ + h * HD_ + d] = (_Float16)out;
}

// ---------------------------------------------------------------------------
extern "C" void kernel_launch(void* const* d_in, const int* in_sizes, int n_in,
                              void* d_out, int out_size, void* d_ws, size_t ws_size,
                              hipStream_t stream) {
    const float* x    = (const float*)d_in[0];
    const float* fcos = (const float*)d_in[1];
    const float* fsin = (const float*)d_in[2];
    const float* cvec = (const float*)d_in[3];
    const float* Wq   = (const float*)d_in[4];
    const float* bq   = (const float*)d_in[5];
    const float* Wk   = (const float*)d_in[6];
    const float* bk   = (const float*)d_in[7];
    const float* Wv   = (const float*)d_in[8];
    const float* bv   = (const float*)d_in[9];
    const float* Wo   = (const float*)d_in[10];
    const float* bo   = (const float*)d_in[11];
    const float* W1   = (const float*)d_in[12];
    const float* b1   = (const float*)d_in[13];
    const float* W2   = (const float*)d_in[14];
    const float* b2   = (const float*)d_in[15];
    const float* ln1s = (const float*)d_in[16];
    const float* ln1b = (const float*)d_in[17];
    const float* ln2s = (const float*)d_in[18];
    const float* ln2b = (const float*)d_in[19];
    const float* geo  = (const float*)d_in[20];
    const float* hash = (const float*)d_in[21];
    float* out = (float*)d_out;

    // bump allocator on d_ws
    size_t off = 0;
    auto alloc = [&](size_t bytes) -> void* {
        void* p = (char*)d_ws + off;
        off = (off + bytes + 255) & ~(size_t)255;
        return p;
    };
    _Float16* xn_h   = (_Float16*)alloc((size_t)M_ * DIM_ * 2);
    _Float16* wq_t   = (_Float16*)alloc((size_t)DIM_ * DIM_ * 2);
    _Float16* wk_t   = (_Float16*)alloc((size_t)DIM_ * DIM_ * 2);
    _Float16* wv_t   = (_Float16*)alloc((size_t)DIM_ * DIM_ * 2);
    _Float16* wo_t   = (_Float16*)alloc((size_t)DIM_ * DIM_ * 2);
    _Float16* w1_t   = (_Float16*)alloc((size_t)DIM_ * FF_ * 2);
    _Float16* w2_t   = (_Float16*)alloc((size_t)FF_ * DIM_ * 2);
    float*    qf     = (float*)alloc((size_t)M_ * DIM_ * 4);
    float*    kf     = (float*)alloc((size_t)M_ * DIM_ * 4);
    float*    vf     = (float*)alloc((size_t)M_ * DIM_ * 4);
    float*    qn     = (float*)alloc((size_t)B_ * N_ * NH_ * 4);
    float*    kn     = (float*)alloc((size_t)B_ * N_ * NH_ * 4);
    _Float16* attn_h = (_Float16*)alloc((size_t)M_ * DIM_ * 2);
    float*    xout   = (float*)alloc((size_t)M_ * DIM_ * 4);
    _Float16* h_h    = (_Float16*)alloc((size_t)M_ * DIM_ * 2);
    _Float16* g_h    = (_Float16*)alloc((size_t)M_ * FF_ * 2);

    // 1. weight transpose + f16 conversion
    convert_w_kernel<<<(DIM_ * DIM_) / 256, 256, 0, stream>>>(Wq, wq_t, DIM_, DIM_);
    convert_w_kernel<<<(DIM_ * DIM_) / 256, 256, 0, stream>>>(Wk, wk_t, DIM_, DIM_);
    convert_w_kernel<<<(DIM_ * DIM_) / 256, 256, 0, stream>>>(Wv, wv_t, DIM_, DIM_);
    convert_w_kernel<<<(DIM_ * DIM_) / 256, 256, 0, stream>>>(Wo, wo_t, DIM_, DIM_);
    convert_w_kernel<<<(DIM_ * FF_) / 256, 256, 0, stream>>>(W1, w1_t, DIM_, FF_);
    convert_w_kernel<<<(FF_ * DIM_) / 256, 256, 0, stream>>>(W2, w2_t, FF_, DIM_);

    // 2. LN1
    layernorm_kernel<<<M_, 256, 0, stream>>>(x, ln1s, ln1b, xn_h);

    // 3. Q/K/V projections (WMMA + async-LDS staged)
    {
        int blocks = (M_ / 128) * (DIM_ / 64);   // 128
        wmma_gemm_kernel<<<blocks, 256, 0, stream>>>(xn_h, wq_t, bq, qf, nullptr, nullptr, DIM_, DIM_, 0);
        wmma_gemm_kernel<<<blocks, 256, 0, stream>>>(xn_h, wk_t, bk, kf, nullptr, nullptr, DIM_, DIM_, 0);
        wmma_gemm_kernel<<<blocks, 256, 0, stream>>>(xn_h, wv_t, bv, vf, nullptr, nullptr, DIM_, DIM_, 0);
    }

    // 4. hash offset + RoPE + expmap0 + norms
    qkv_post_kernel<<<(B_ * N_ * NH_) / 8, 256, 0, stream>>>(qf, kf, fcos, fsin, cvec, hash, qn, kn);

    // 5. sliding-window Poincare attention -> f16
    attention_kernel<<<(B_ * NH_ * N_) / 4, 256, 0, stream>>>(qf, kf, vf, qn, kn, cvec, geo, attn_h);

    // 6. output projection with fused residual: xout = x + attn@Wo + bo
    wmma_gemm_kernel<<<(M_ / 128) * (DIM_ / 64), 256, 0, stream>>>(attn_h, wo_t, bo, xout, nullptr, x, DIM_, DIM_, 2);

    // 7. LN2
    layernorm_kernel<<<M_, 256, 0, stream>>>(xout, ln2s, ln2b, h_h);

    // 8. FFN up + GELU (f16 out)
    wmma_gemm_kernel<<<(M_ / 128) * (FF_ / 64), 256, 0, stream>>>(h_h, w1_t, b1, nullptr, g_h, nullptr, FF_, DIM_, 1);

    // 9. FFN down with fused residual -> final output
    wmma_gemm_kernel<<<(M_ / 128) * (DIM_ / 64), 256, 0, stream>>>(g_h, w2_t, b2, out, nullptr, xout, DIM_, FF_, 2);

    (void)in_sizes; (void)n_in; (void)out_size; (void)ws_size;
}
